// MPNN_md17_2774548873305
// MI455X (gfx1250) — matrix operand order repaired and checked by reading, not compiled, not run on hardware
//
#include <hip/hip_runtime.h>
#include <math.h>

// ---------------------------------------------------------------------------
// MPNN (MD17) on MI455X / gfx1250.
// All GEMMs use v_wmma_f32_16x16x32_bf16 (bf16 in, f32 accumulate).
// Weights converted once per launch to bf16 transposed [n][Kpad] layout.
// Activations kept bf16 (8MB) -> fully L2 resident; agg kept f32 (16MB).
// Scatter-add of edge messages uses global_atomic_add_f32 (L2 atomic units).
// Edge/update kernels run TWO 16-row M-tiles per wave: two independent WMMA
// accumulator chains share every B fragment (halves LDS traffic, keeps the
// XDL pipe busy while the other chain's operands land).
// ---------------------------------------------------------------------------

typedef unsigned short u16;
typedef unsigned int   u32;
typedef __attribute__((ext_vector_type(16))) __bf16 bf16x16;
typedef __attribute__((ext_vector_type(8)))  float  f32x8;
typedef __attribute__((ext_vector_type(4)))  float  f32x4;
typedef __attribute__((ext_vector_type(4)))  u32    u32x4;
typedef __attribute__((ext_vector_type(2)))  u32    u32x2;

union FragU { bf16x16 f; u32x4 q[2]; };

__device__ __forceinline__ u16 f2bf(float x) {
  union { __bf16 b; u16 s; } u;
  u.b = (__bf16)x;          // compiler picks hw cvt (RNE) if available
  return u.s;
}

// A fragment (16x32 bf16, ISA 7.12.2): lane row = lane&15, hi = lane>>4.
// elems 0..7  = A[row][k0 + hi*8 .. +7], elems 8..15 = A[row][k0+16+hi*8 .. +7]
__device__ __forceinline__ void load_a_row(const u16* rowptr, int k0, int hi, FragU& a) {
  const u32x4* p = (const u32x4*)(rowptr + k0 + (hi << 3));
  a.q[0] = p[0];
  a.q[1] = p[2];
}

// B fragment (32x16 bf16): lane col = lane&15, hi = lane>>4; holds
// W[k0 + hi*8 .. +7][col] and W[k0+16+hi*8 .. +7][col], stored transposed in LDS as wt[col*KL + k].
__device__ __forceinline__ void load_b(const u16* wt, int KL, int col, int k0, int hi, FragU& b) {
  const u32x4* p = (const u32x4*)(wt + (size_t)col * KL + k0 + (hi << 3));
  b.q[0] = p[0];
  b.q[1] = p[2];
}

__device__ __forceinline__ f32x8 wmma_bf(const FragU& a, const FragU& b, f32x8 c) {
  return __builtin_amdgcn_wmma_f32_16x16x32_bf16(false, a.f, false, b.f, (short)0, c, false, false);
}

// cooperative copy of a [rows][K] bf16 weight into LDS with row stride KL
__device__ __forceinline__ void copy_w(const u16* g, u16* l, int rows, int K, int KL,
                                       int tid, int nthr) {
  int chunks = K >> 3;  // u32x4 (8 bf16) per chunk
  for (int i = tid; i < rows * chunks; i += nthr) {
    int r = i / chunks, q = i - r * chunks;
    ((u32x4*)(l + r * KL))[q] = ((const u32x4*)(g + (size_t)r * K))[q];
  }
}

// ---------------- constants ----------------
static constexpr int NN = 32768;     // nodes
static constexpr int NE = 524288;    // edges
static constexpr int NB = 2048;      // batch
static constexpr int KL256 = 264;    // LDS strides (bank-conflict padding)
static constexpr int KL128 = 136;
static constexpr int KL64  = 72;
static constexpr int HS    = 136;    // hidden tile stride

// ---------------- weight conversion: W[k][n] f32 -> Wt[n][kpad] bf16 --------
__global__ void cvt_weight(const float* __restrict__ W, u16* __restrict__ Wt,
                           int K, int Nout, int Kpad, int NoutPad) {
  int total = NoutPad * Kpad;
  for (int i = blockIdx.x * blockDim.x + threadIdx.x; i < total;
       i += gridDim.x * blockDim.x) {
    int n = i / Kpad, k = i - n * Kpad;
    float v = (k < K && n < Nout) ? W[(size_t)k * Nout + n] : 0.0f;
    Wt[i] = f2bf(v);
  }
}

// ---------------- embedder: (N,60) -> relu(.W1+b1).W2+b2 -> x0 bf16 --------
__global__ __launch_bounds__(128) void embed_kernel(
    const float* __restrict__ loc, const float* __restrict__ vel,
    const u16* __restrict__ wt1g, const float* __restrict__ b1,
    const u16* __restrict__ wt2g, const float* __restrict__ b2,
    u16* __restrict__ xout, int ntiles) {
  extern __shared__ char smem_raw[];
  u16* wt1 = (u16*)smem_raw;            // [128][72]
  u16* wt2 = wt1 + 128 * KL64;          // [128][136]
  int tid = threadIdx.x, wave = tid >> 5, lane = tid & 31;
  int lanelo = lane & 15, hi = lane >> 4;
  u16* aT  = wt2 + 128 * KL128 + wave * (16 * KL64);
  u16* hid = wt2 + 128 * KL128 + 4 * (16 * KL64) + wave * (16 * HS);

  copy_w(wt1g, wt1, 128, 64, KL64, tid, 128);
  copy_w(wt2g, wt2, 128, 128, KL128, tid, 128);
  __syncthreads();

  for (int t = blockIdx.x * 4 + wave; t < ntiles; t += gridDim.x * 4) {
    int nbase = t * 16;
    // stage interleaved (loc,vel) rows as bf16 [16][64] (cols 60..63 zero)
    for (int i = lane; i < 16 * 64; i += 32) {
      int row = i >> 6, c = i & 63;
      float v = 0.0f;
      if (c < 60) {
        int f = c / 6, j = c - f * 6, n = nbase + row;
        v = (j < 3) ? loc[(size_t)n * 30 + f * 3 + j]
                    : vel[(size_t)n * 30 + f * 3 + (j - 3)];
      }
      aT[row * KL64 + c] = f2bf(v);
    }
    __builtin_amdgcn_wave_barrier();
    FragU a[2];
    load_a_row(aT + lanelo * KL64, 0, hi, a[0]);
    load_a_row(aT + lanelo * KL64, 32, hi, a[1]);
#pragma unroll
    for (int ct = 0; ct < 8; ct++) {
      int col = ct * 16 + lanelo;
      FragU bb0, bb1;
      load_b(wt1, KL64, col, 0, hi, bb0);
      load_b(wt1, KL64, col, 32, hi, bb1);
      f32x8 acc = {0, 0, 0, 0, 0, 0, 0, 0};
      acc = wmma_bf(a[0], bb0, acc);
      acc = wmma_bf(a[1], bb1, acc);
      float biasv = b1[col];
#pragma unroll
      for (int r = 0; r < 8; r++)
        hid[(r + hi * 8) * HS + col] = f2bf(fmaxf(acc[r] + biasv, 0.0f));
    }
    __builtin_amdgcn_wave_barrier();
    FragU a2[4];
#pragma unroll
    for (int kt = 0; kt < 4; kt++) load_a_row(hid + lanelo * HS, kt * 32, hi, a2[kt]);
#pragma unroll
    for (int ct = 0; ct < 8; ct++) {
      int col = ct * 16 + lanelo;
      FragU bb[4];
#pragma unroll
      for (int kt = 0; kt < 4; kt++) load_b(wt2, KL128, col, kt * 32, hi, bb[kt]);
      f32x8 acc = {0, 0, 0, 0, 0, 0, 0, 0};
#pragma unroll
      for (int kt = 0; kt < 4; kt++) acc = wmma_bf(a2[kt], bb[kt], acc);
      float biasv = b2[col];
#pragma unroll
      for (int r = 0; r < 8; r++)
        xout[(size_t)(nbase + r + hi * 8) * 128 + col] = f2bf(acc[r] + biasv);
    }
    __builtin_amdgcn_wave_barrier();
  }
}

// ---------------- edge messages: msg MLP + atomic scatter-add --------------
// Two 16-edge M-tiles per wave; both WMMA chains share each B fragment.
__global__ __launch_bounds__(256) void edge_kernel(
    const u16* __restrict__ xin,
    const u16* __restrict__ wt1g, const float* __restrict__ b1,
    const u16* __restrict__ wt2g, const float* __restrict__ b2,
    const int* __restrict__ src, const int* __restrict__ dst,
    float* __restrict__ agg, int npairs) {
  extern __shared__ char smem_raw[];
  u16* wt1 = (u16*)smem_raw;            // [128][264]
  u16* wt2 = wt1 + 128 * KL256;         // [128][136]
  int tid = threadIdx.x, wave = tid >> 5, lane = tid & 31;
  int lanelo = lane & 15, hi = lane >> 4;
  u16* hid = wt2 + 128 * KL128 + wave * (2 * 16 * HS);  // [32][136]

  copy_w(wt1g, wt1, 128, 256, KL256, tid, 256);
  copy_w(wt2g, wt2, 128, 128, KL128, tid, 256);
  __syncthreads();

  for (int t = blockIdx.x * 8 + wave; t < npairs; t += gridDim.x * 8) {
    int ebase = t * 32;
    int e0 = ebase + lanelo, e1 = ebase + 16 + lanelo;
    // A fragments gathered straight from bf16 x rows (K contiguous in memory)
    FragU a0[8], a1[8];
    {
      const u16* rd0 = xin + (size_t)dst[e0] * 128;
      const u16* rs0 = xin + (size_t)src[e0] * 128;
      const u16* rd1 = xin + (size_t)dst[e1] * 128;
      const u16* rs1 = xin + (size_t)src[e1] * 128;
#pragma unroll
      for (int kt = 0; kt < 4; kt++) {
        load_a_row(rd0, kt * 32, hi, a0[kt]);
        load_a_row(rs0, kt * 32, hi, a0[4 + kt]);
        load_a_row(rd1, kt * 32, hi, a1[kt]);
        load_a_row(rs1, kt * 32, hi, a1[4 + kt]);
      }
    }
#pragma unroll
    for (int ct = 0; ct < 8; ct++) {
      int col = ct * 16 + lanelo;
      f32x8 acc0 = {0, 0, 0, 0, 0, 0, 0, 0};
      f32x8 acc1 = {0, 0, 0, 0, 0, 0, 0, 0};
#pragma unroll
      for (int kt = 0; kt < 8; kt++) {
        FragU b; load_b(wt1, KL256, col, kt * 32, hi, b);
        acc0 = wmma_bf(a0[kt], b, acc0);
        acc1 = wmma_bf(a1[kt], b, acc1);
      }
      float biasv = b1[col];
#pragma unroll
      for (int r = 0; r < 8; r++) {
        hid[(r + hi * 8) * HS + col]      = f2bf(fmaxf(acc0[r] + biasv, 0.0f));
        hid[(16 + r + hi * 8) * HS + col] = f2bf(fmaxf(acc1[r] + biasv, 0.0f));
      }
    }
    __builtin_amdgcn_wave_barrier();
    FragU a20[4], a21[4];
#pragma unroll
    for (int kt = 0; kt < 4; kt++) {
      load_a_row(hid + lanelo * HS, kt * 32, hi, a20[kt]);
      load_a_row(hid + (16 + lanelo) * HS, kt * 32, hi, a21[kt]);
    }
    int drow0[8], drow1[8];
#pragma unroll
    for (int r = 0; r < 8; r++) {
      drow0[r] = dst[ebase + r + hi * 8];
      drow1[r] = dst[ebase + 16 + r + hi * 8];
    }
#pragma unroll
    for (int ct = 0; ct < 8; ct++) {
      int col = ct * 16 + lanelo;
      FragU bb[4];
#pragma unroll
      for (int kt = 0; kt < 4; kt++) load_b(wt2, KL128, col, kt * 32, hi, bb[kt]);
      f32x8 acc0 = {0, 0, 0, 0, 0, 0, 0, 0};
      f32x8 acc1 = {0, 0, 0, 0, 0, 0, 0, 0};
#pragma unroll
      for (int kt = 0; kt < 4; kt++) {
        acc0 = wmma_bf(a20[kt], bb[kt], acc0);
        acc1 = wmma_bf(a21[kt], bb[kt], acc1);
      }
      float biasv = b2[col];
#pragma unroll
      for (int r = 0; r < 8; r++) {
        atomicAdd(&agg[(size_t)drow0[r] * 128 + col], acc0[r] + biasv);
        atomicAdd(&agg[(size_t)drow1[r] * 128 + col], acc1[r] + biasv);
      }
    }
    __builtin_amdgcn_wave_barrier();
  }
}

// ---------------- node update: relu(cat(x,agg).W1+b1).W2+b2 -> x_new -------
// Two 16-node M-tiles per wave, shared B fragments.
__global__ __launch_bounds__(256) void update_kernel(
    const u16* __restrict__ xin, const float* __restrict__ agg,
    const u16* __restrict__ wt1g, const float* __restrict__ b1,
    const u16* __restrict__ wt2g, const float* __restrict__ b2,
    u16* __restrict__ xout, int npairs) {
  extern __shared__ char smem_raw[];
  u16* wt1 = (u16*)smem_raw;            // [128][264]
  u16* wt2 = wt1 + 128 * KL256;         // [128][136]
  int tid = threadIdx.x, wave = tid >> 5, lane = tid & 31;
  int lanelo = lane & 15, hi = lane >> 4;
  u16* aggL = wt2 + 128 * KL128 + wave * (2 * 16 * HS);                   // [32][136]
  u16* hid  = wt2 + 128 * KL128 + 8 * (2 * 16 * HS) + wave * (2 * 16 * HS);

  copy_w(wt1g, wt1, 128, 256, KL256, tid, 256);
  copy_w(wt2g, wt2, 128, 128, KL128, tid, 256);
  __syncthreads();

  for (int t = blockIdx.x * 8 + wave; t < npairs; t += gridDim.x * 8) {
    int nbase = t * 32;
    // stage 32 agg rows f32 -> bf16
    for (int i = lane; i < 32 * 32; i += 32) {
      int row = i >> 5, c4 = i & 31;
      f32x4 v = ((const f32x4*)(agg + (size_t)(nbase + row) * 128))[c4];
      u32x2 pk;
      pk.x = (u32)f2bf(v.x) | ((u32)f2bf(v.y) << 16);
      pk.y = (u32)f2bf(v.z) | ((u32)f2bf(v.w) << 16);
      *((u32x2*)(aggL + row * HS + c4 * 4)) = pk;
    }
    __builtin_amdgcn_wave_barrier();
    FragU a0[8], a1[8];
    {
      const u16* rx0 = xin + (size_t)(nbase + lanelo) * 128;
      const u16* rx1 = xin + (size_t)(nbase + 16 + lanelo) * 128;
#pragma unroll
      for (int kt = 0; kt < 4; kt++) {
        load_a_row(rx0, kt * 32, hi, a0[kt]);
        load_a_row(aggL + lanelo * HS, kt * 32, hi, a0[4 + kt]);
        load_a_row(rx1, kt * 32, hi, a1[kt]);
        load_a_row(aggL + (16 + lanelo) * HS, kt * 32, hi, a1[4 + kt]);
      }
    }
#pragma unroll
    for (int ct = 0; ct < 8; ct++) {
      int col = ct * 16 + lanelo;
      f32x8 acc0 = {0, 0, 0, 0, 0, 0, 0, 0};
      f32x8 acc1 = {0, 0, 0, 0, 0, 0, 0, 0};
#pragma unroll
      for (int kt = 0; kt < 8; kt++) {
        FragU b; load_b(wt1, KL256, col, kt * 32, hi, b);
        acc0 = wmma_bf(a0[kt], b, acc0);
        acc1 = wmma_bf(a1[kt], b, acc1);
      }
      float biasv = b1[col];
#pragma unroll
      for (int r = 0; r < 8; r++) {
        hid[(r + hi * 8) * HS + col]      = f2bf(fmaxf(acc0[r] + biasv, 0.0f));
        hid[(16 + r + hi * 8) * HS + col] = f2bf(fmaxf(acc1[r] + biasv, 0.0f));
      }
    }
    __builtin_amdgcn_wave_barrier();
    FragU a20[4], a21[4];
#pragma unroll
    for (int kt = 0; kt < 4; kt++) {
      load_a_row(hid + lanelo * HS, kt * 32, hi, a20[kt]);
      load_a_row(hid + (16 + lanelo) * HS, kt * 32, hi, a21[kt]);
    }
#pragma unroll
    for (int ct = 0; ct < 8; ct++) {
      int col = ct * 16 + lanelo;
      FragU bb[4];
#pragma unroll
      for (int kt = 0; kt < 4; kt++) load_b(wt2, KL128, col, kt * 32, hi, bb[kt]);
      f32x8 acc0 = {0, 0, 0, 0, 0, 0, 0, 0};
      f32x8 acc1 = {0, 0, 0, 0, 0, 0, 0, 0};
#pragma unroll
      for (int kt = 0; kt < 4; kt++) {
        acc0 = wmma_bf(a20[kt], bb[kt], acc0);
        acc1 = wmma_bf(a21[kt], bb[kt], acc1);
      }
      float biasv = b2[col];
#pragma unroll
      for (int r = 0; r < 8; r++) {
        xout[(size_t)(nbase + r + hi * 8) * 128 + col]      = f2bf(acc0[r] + biasv);
        xout[(size_t)(nbase + 16 + r + hi * 8) * 128 + col] = f2bf(acc1[r] + biasv);
      }
    }
    __builtin_amdgcn_wave_barrier();
  }
}

// ---------------- head: x -> 30 outputs, per-node ADE/FDE ------------------
__global__ __launch_bounds__(128) void head_kernel(
    const u16* __restrict__ xin,
    const u16* __restrict__ wt1g, const float* __restrict__ b1,
    const u16* __restrict__ wt2g, const float* __restrict__ b2,
    const float* __restrict__ loc, const float* __restrict__ y,
    float* __restrict__ node_ade, float* __restrict__ node_fde, int ntiles) {
  extern __shared__ char smem_raw[];
  u16* wt1 = (u16*)smem_raw;            // [128][136]
  u16* wt2 = wt1 + 128 * KL128;         // [32][136]
  int tid = threadIdx.x, wave = tid >> 5, lane = tid & 31;
  int lanelo = lane & 15, hi = lane >> 4;
  u16* hid = wt2 + 32 * KL128 + wave * (16 * HS);
  float* outt = (float*)(wt2 + 32 * KL128 + 4 * (16 * HS)) + wave * (16 * 32);

  copy_w(wt1g, wt1, 128, 128, KL128, tid, 128);
  copy_w(wt2g, wt2, 32, 128, KL128, tid, 128);
  __syncthreads();

  for (int t = blockIdx.x * 4 + wave; t < ntiles; t += gridDim.x * 4) {
    int nbase = t * 16;
    FragU a[4];
    const u16* rx = xin + (size_t)(nbase + lanelo) * 128;
#pragma unroll
    for (int kt = 0; kt < 4; kt++) load_a_row(rx, kt * 32, hi, a[kt]);
#pragma unroll
    for (int ct = 0; ct < 8; ct++) {
      int col = ct * 16 + lanelo;
      FragU bb[4];
#pragma unroll
      for (int kt = 0; kt < 4; kt++) load_b(wt1, KL128, col, kt * 32, hi, bb[kt]);
      f32x8 acc = {0, 0, 0, 0, 0, 0, 0, 0};
#pragma unroll
      for (int kt = 0; kt < 4; kt++) acc = wmma_bf(a[kt], bb[kt], acc);
      float biasv = b1[col];
#pragma unroll
      for (int r = 0; r < 8; r++)
        hid[(r + hi * 8) * HS + col] = f2bf(fmaxf(acc[r] + biasv, 0.0f));
    }
    __builtin_amdgcn_wave_barrier();
    FragU a2[4];
#pragma unroll
    for (int kt = 0; kt < 4; kt++) load_a_row(hid + lanelo * HS, kt * 32, hi, a2[kt]);
#pragma unroll
    for (int ct = 0; ct < 2; ct++) {
      int col = ct * 16 + lanelo;
      FragU bb[4];
#pragma unroll
      for (int kt = 0; kt < 4; kt++) load_b(wt2, KL128, col, kt * 32, hi, bb[kt]);
      f32x8 acc = {0, 0, 0, 0, 0, 0, 0, 0};
#pragma unroll
      for (int kt = 0; kt < 4; kt++) acc = wmma_bf(a2[kt], bb[kt], acc);
      float biasv = (col < 30) ? b2[col] : 0.0f;
#pragma unroll
      for (int r = 0; r < 8; r++)
        outt[(r + hi * 8) * 32 + col] = acc[r] + biasv;
    }
    __builtin_amdgcn_wave_barrier();
    if (lane < 16) {
      int n = nbase + lane;
      float asum = 0.0f, fd = 0.0f;
#pragma unroll
      for (int f = 0; f < 10; f++) {
        float dx = outt[lane * 32 + f * 3 + 0] + loc[(size_t)n * 30 + f * 3 + 0] - y[(size_t)n * 30 + f * 3 + 0];
        float dy = outt[lane * 32 + f * 3 + 1] + loc[(size_t)n * 30 + f * 3 + 1] - y[(size_t)n * 30 + f * 3 + 1];
        float dz = outt[lane * 32 + f * 3 + 2] + loc[(size_t)n * 30 + f * 3 + 2] - y[(size_t)n * 30 + f * 3 + 2];
        float d = sqrtf(dx * dx + dy * dy + dz * dz);
        asum += d;
        if (f == 9) fd = d;
      }
      node_ade[n] = asum;
      node_fde[n] = fd;
    }
    __builtin_amdgcn_wave_barrier();
  }
}

// ---------------- batch reductions ----------------------------------------
__global__ void batch_kernel(const float* __restrict__ node_ade,
                             const float* __restrict__ node_fde,
                             float* __restrict__ out) {
  int b = blockIdx.x * blockDim.x + threadIdx.x;
  if (b < NB) {
    float a = 0.f, f = 0.f;
#pragma unroll
    for (int i = 0; i < 16; i++) {
      a += node_ade[b * 16 + i];
      f += node_fde[b * 16 + i];
    }
    out[1 + b] = a * (1.0f / 160.0f);       // mean over 16 nodes * 10 frames
    out[1 + NB + b] = f * (1.0f / 16.0f);   // mean over 16 nodes
  }
}

__global__ void final_kernel(float* __restrict__ out) {
  __shared__ float s[256];
  float acc = 0.f;
  for (int i = threadIdx.x; i < NB; i += 256) acc += out[1 + i];
  s[threadIdx.x] = acc;
  __syncthreads();
  for (int w = 128; w > 0; w >>= 1) {
    if ((int)threadIdx.x < w) s[threadIdx.x] += s[threadIdx.x + w];
    __syncthreads();
  }
  if (threadIdx.x == 0) out[0] = s[0] / (float)NB;
}

// ---------------- workspace layout ----------------------------------------
static constexpr size_t WS_WT_EMB1 = 0;                                   // [128][64]
static constexpr size_t WS_WT_EMB2 = WS_WT_EMB1 + 128 * 64 * 2;           // [128][128]
static constexpr size_t WS_WT_MSG1 = WS_WT_EMB2 + 128 * 128 * 2;          // 7x[128][256]
static constexpr size_t WS_WT_MSG2 = WS_WT_MSG1 + 7ull * 128 * 256 * 2;   // 7x[128][128]
static constexpr size_t WS_WT_UPD1 = WS_WT_MSG2 + 7ull * 128 * 128 * 2;
static constexpr size_t WS_WT_UPD2 = WS_WT_UPD1 + 7ull * 128 * 256 * 2;
static constexpr size_t WS_WT_HD1  = WS_WT_UPD2 + 7ull * 128 * 128 * 2;
static constexpr size_t WS_WT_HD2  = WS_WT_HD1 + 128 * 128 * 2;
static constexpr size_t WS_X0      = WS_WT_HD2 + 32 * 128 * 2;
static constexpr size_t WS_X1      = WS_X0 + (size_t)NN * 128 * 2;
static constexpr size_t WS_AGG     = WS_X1 + (size_t)NN * 128 * 2;
static constexpr size_t WS_ADE     = WS_AGG + (size_t)NN * 128 * 4;
static constexpr size_t WS_FDE     = WS_ADE + (size_t)NN * 4;

// dynamic LDS sizes (bytes)
static constexpr size_t LDS_EMB  = (size_t)(128 * KL64 + 128 * KL128 + 4 * 16 * KL64 + 4 * 16 * HS) * 2;
static constexpr size_t LDS_EDGE = (size_t)(128 * KL256 + 128 * KL128 + 8 * 2 * 16 * HS) * 2;
static constexpr size_t LDS_UPD  = (size_t)(128 * KL256 + 128 * KL128 + 16 * 2 * 16 * HS) * 2;
static constexpr size_t LDS_HEAD = (size_t)(128 * KL128 + 32 * KL128 + 4 * 16 * HS) * 2 + 4 * 16 * 32 * 4;

extern "C" void kernel_launch(void* const* d_in, const int* in_sizes, int n_in,
                              void* d_out, int out_size, void* d_ws, size_t ws_size,
                              hipStream_t stream) {
  (void)in_sizes; (void)n_in; (void)out_size; (void)ws_size;
  const float* loc     = (const float*)d_in[0];
  const float* vel     = (const float*)d_in[1];
  const float* y       = (const float*)d_in[2];
  const int*   ei      = (const int*)d_in[3];
  // d_in[4] = batch_size (fixed = 2048)
  const float* emb_W1  = (const float*)d_in[5];
  const float* emb_b1  = (const float*)d_in[6];
  const float* emb_W2  = (const float*)d_in[7];
  const float* emb_b2  = (const float*)d_in[8];
  const float* msg_W1  = (const float*)d_in[9];
  const float* msg_b1  = (const float*)d_in[10];
  const float* msg_W2  = (const float*)d_in[11];
  const float* msg_b2  = (const float*)d_in[12];
  const float* upd_W1  = (const float*)d_in[13];
  const float* upd_b1  = (const float*)d_in[14];
  const float* upd_W2  = (const float*)d_in[15];
  const float* upd_b2  = (const float*)d_in[16];
  const float* head_W1 = (const float*)d_in[17];
  const float* head_b1 = (const float*)d_in[18];
  const float* head_W2 = (const float*)d_in[19];
  const float* head_b2 = (const float*)d_in[20];

  char* ws = (char*)d_ws;
  u16*   wtE1 = (u16*)(ws + WS_WT_EMB1);
  u16*   wtE2 = (u16*)(ws + WS_WT_EMB2);
  u16*   wtM1 = (u16*)(ws + WS_WT_MSG1);
  u16*   wtM2 = (u16*)(ws + WS_WT_MSG2);
  u16*   wtU1 = (u16*)(ws + WS_WT_UPD1);
  u16*   wtU2 = (u16*)(ws + WS_WT_UPD2);
  u16*   wtH1 = (u16*)(ws + WS_WT_HD1);
  u16*   wtH2 = (u16*)(ws + WS_WT_HD2);
  u16*   xb[2] = { (u16*)(ws + WS_X0), (u16*)(ws + WS_X1) };
  float* aggp  = (float*)(ws + WS_AGG);
  float* adep  = (float*)(ws + WS_ADE);
  float* fdep  = (float*)(ws + WS_FDE);
  float* out   = (float*)d_out;

  const int* src = ei;
  const int* dst = ei + NE;

  // ---- weight conversion (f32 -> bf16 transposed [n][Kpad]) ----
  cvt_weight<<<32, 256, 0, stream>>>(emb_W1, wtE1, 60, 128, 64, 128);
  cvt_weight<<<64, 256, 0, stream>>>(emb_W2, wtE2, 128, 128, 128, 128);
  for (int i = 0; i < 7; i++) {
    cvt_weight<<<128, 256, 0, stream>>>(msg_W1 + (size_t)i * 256 * 128,
                                        wtM1 + (size_t)i * 128 * 256, 256, 128, 256, 128);
    cvt_weight<<<64, 256, 0, stream>>>(msg_W2 + (size_t)i * 128 * 128,
                                       wtM2 + (size_t)i * 128 * 128, 128, 128, 128, 128);
    cvt_weight<<<128, 256, 0, stream>>>(upd_W1 + (size_t)i * 256 * 128,
                                        wtU1 + (size_t)i * 128 * 256, 256, 128, 256, 128);
    cvt_weight<<<64, 256, 0, stream>>>(upd_W2 + (size_t)i * 128 * 128,
                                       wtU2 + (size_t)i * 128 * 128, 128, 128, 128, 128);
  }
  cvt_weight<<<64, 256, 0, stream>>>(head_W1, wtH1, 128, 128, 128, 128);
  cvt_weight<<<16, 256, 0, stream>>>(head_W2, wtH2, 128, 30, 128, 32);

  // ---- embedder ----
  embed_kernel<<<256, 128, LDS_EMB, stream>>>(loc, vel, wtE1, emb_b1, wtE2, emb_b2,
                                              xb[0], NN / 16);

  // ---- message passing layers ----
  for (int i = 0; i < 7; i++) {
    hipMemsetAsync(aggp, 0, (size_t)NN * 128 * 4, stream);
    edge_kernel<<<512, 256, LDS_EDGE, stream>>>(
        xb[i & 1], wtM1 + (size_t)i * 128 * 256, msg_b1 + i * 128,
        wtM2 + (size_t)i * 128 * 128, msg_b2 + i * 128, src, dst, aggp, NE / 32);
    update_kernel<<<128, 256, LDS_UPD, stream>>>(
        xb[i & 1], aggp, wtU1 + (size_t)i * 128 * 256, upd_b1 + i * 128,
        wtU2 + (size_t)i * 128 * 128, upd_b2 + i * 128, xb[(i + 1) & 1], NN / 32);
  }

  // ---- head + losses ----
  head_kernel<<<256, 128, LDS_HEAD, stream>>>(xb[1], wtH1, head_b1, wtH2, head_b2,
                                              loc, y, adep, fdep, NN / 16);
  batch_kernel<<<(NB + 255) / 256, 256, 0, stream>>>(adep, fdep, out);
  final_kernel<<<1, 256, 0, stream>>>(out);
}
